// GNN_Model_56186762166752
// MI455X (gfx1250) — matrix operand order repaired and checked by reading, not compiled, not run on hardware
//
#include <hip/hip_runtime.h>
#include <hip/hip_bf16.h>

typedef __attribute__((ext_vector_type(16))) _Float16 v16h;
typedef __attribute__((ext_vector_type(8)))  _Float16 v8h;
typedef __attribute__((ext_vector_type(4)))  _Float16 v4h;
typedef __attribute__((ext_vector_type(8)))  float    v8f;

#define N_NODES 50000
#define N_EDGES 1600000
#define NPAD    50048          // 391 * 128
#define HID     128
#define WSTR    136            // padded half-stride: 272B (16B aligned, bank-conflict free)

static __host__ __device__ inline int cdiv(int a, int b) { return (a + b - 1) / b; }

// ---------------------------------------------------------------------------
// Weight conversion: wp[l][hop][n][k] (f16, transposed) from W_in / W_hid
// ---------------------------------------------------------------------------
__global__ void k_cvt_weights(const float* __restrict__ Win,
                              const float* __restrict__ Whid,
                              _Float16* __restrict__ wp) {
    int idx = blockIdx.x * 256 + threadIdx.x;
    if (idx >= 8 * 2 * HID * HID) return;
    int l   = idx >> 15;           // / 32768
    int rem = idx & 32767;
    int hop = rem >> 14;
    int nk  = rem & 16383;
    int n = nk >> 7, k = nk & 127;
    float v;
    if (l == 0) v = (k < 12) ? Win[(hop * 12 + k) * HID + n] : 0.0f;
    else        v = Whid[(((l - 1) * 2 + hop) * HID + k) * HID + n];
    wp[idx] = (_Float16)v;
}

// ---------------------------------------------------------------------------
// Initial node features: [x_t | t_emb[t[i/5000]] | zeros] -> f16, zero padded
// ---------------------------------------------------------------------------
__global__ void k_embed(const float* __restrict__ x_t, const int* __restrict__ t,
                        const float* __restrict__ temb, _Float16* __restrict__ h16) {
    int idx = blockIdx.x * 256 + threadIdx.x;
    if (idx >= NPAD * HID) return;
    int i = idx >> 7, c = idx & 127;
    _Float16 v = (_Float16)0.0f;
    if (i < N_NODES) {
        if (c < 2) v = (_Float16)x_t[i * 2 + c];
        else if (c < 12) {
            int g = i / 5000;
            v = (_Float16)temb[t[g] * 10 + (c - 2)];
        }
    }
    h16[idx] = v;
}

// ---------------------------------------------------------------------------
// CSR build: count -> dinv -> scan -> fill
// ---------------------------------------------------------------------------
__global__ void k_zero(int* __restrict__ cnt) {
    int i = blockIdx.x * 256 + threadIdx.x;
    if (i < NPAD) cnt[i] = 0;
}
__global__ void k_count(const int* __restrict__ ei, int* __restrict__ cnt) {
    int e = blockIdx.x * 256 + threadIdx.x;
    if (e < N_EDGES) atomicAdd(&cnt[ei[N_EDGES + e]], 1);
}
__global__ void k_dinv(const int* __restrict__ cnt, float* __restrict__ dinv) {
    int i = blockIdx.x * 256 + threadIdx.x;
    if (i < NPAD) dinv[i] = rsqrtf(1.0f + (float)cnt[i]);   // +1 self loop
}
// single-block exclusive scan over NPAD counts (one-time, tiny)
__global__ __launch_bounds__(1024)
void k_scan(const int* __restrict__ cnt, int* __restrict__ rowptr) {
    __shared__ int tmp[1024];
    __shared__ int carry_s;
    int tid = threadIdx.x;
    if (tid == 0) carry_s = 0;
    __syncthreads();
    for (int base = 0; base < NPAD; base += 1024) {
        int i = base + tid;
        int v = (i < NPAD) ? cnt[i] : 0;
        tmp[tid] = v;
        __syncthreads();
        for (int off = 1; off < 1024; off <<= 1) {
            int t = (tid >= off) ? tmp[tid - off] : 0;
            __syncthreads();
            tmp[tid] += t;
            __syncthreads();
        }
        int incl = tmp[tid];
        int carry = carry_s;
        if (i < NPAD) rowptr[i] = carry + incl - v;
        __syncthreads();
        if (tid == 1023) carry_s = carry + incl;
        __syncthreads();
    }
    if (tid == 0) rowptr[NPAD] = carry_s;
}
__global__ void k_copy(const int* __restrict__ src, int* __restrict__ dst) {
    int i = blockIdx.x * 256 + threadIdx.x;
    if (i < NPAD) dst[i] = src[i];
}
__global__ void k_fill(const int* __restrict__ ei, const float* __restrict__ dinv,
                       int* __restrict__ cursor, int* __restrict__ csr_src,
                       float* __restrict__ csr_w) {
    int e = blockIdx.x * 256 + threadIdx.x;
    if (e >= N_EDGES) return;
    int src = ei[e];
    int dst = ei[N_EDGES + e];
    int pos = atomicAdd(&cursor[dst], 1);
    csr_src[pos] = src;
    csr_w[pos]   = dinv[src] * dinv[dst];
}

// ---------------------------------------------------------------------------
// Gather aggregation (no atomics): agg16[i] = dinv[i]^2*h16[i] + sum_e w_e*h16[src_e]
// (1<<shift) lanes per node, 4 channels per lane; f32 accumulation in registers.
// ---------------------------------------------------------------------------
__global__ void k_gather(const int* __restrict__ rowptr, const int* __restrict__ csr_src,
                         const float* __restrict__ csr_w, const _Float16* __restrict__ h16,
                         const float* __restrict__ dinv, _Float16* __restrict__ agg16,
                         int shift) {
    int lane = threadIdx.x & 31;
    int node = blockIdx.x * (256 >> shift) + (threadIdx.x >> shift);
    if (node >= NPAD) return;
    int c = (lane & ((1 << shift) - 1)) << 2;

    float d = dinv[node];
    v4h hv = *(const v4h*)(h16 + (size_t)node * HID + c);
    float s = d * d;
    float a0 = s * (float)hv[0], a1 = s * (float)hv[1];
    float a2 = s * (float)hv[2], a3 = s * (float)hv[3];

    int beg = rowptr[node], end = rowptr[node + 1];
    for (int j = beg; j < end; ++j) {
        int   src = csr_src[j];
        float w   = csr_w[j];
        v4h sv = *(const v4h*)(h16 + (size_t)src * HID + c);
        a0 += w * (float)sv[0]; a1 += w * (float)sv[1];
        a2 += w * (float)sv[2]; a3 += w * (float)sv[3];
    }
    v4h o;
    o[0] = (_Float16)a0; o[1] = (_Float16)a1;
    o[2] = (_Float16)a2; o[3] = (_Float16)a3;
    *(v4h*)(agg16 + (size_t)node * HID + c) = o;
}

// ---------------------------------------------------------------------------
// WMMA layer: h = relu(h@W0 + b0 + agg@W1 + b1), in-place, 128 rows per block.
// B fragments double-buffered, A prefetched one k-step ahead.
// ---------------------------------------------------------------------------
__device__ inline v16h load_b_frag(const _Float16* __restrict__ w, int nt, int ks,
                                   int ml, int g) {
    const _Float16* pb = w + (nt * 16 + ml) * WSTR + ks + g * 8;
    v8h blo = *(const v8h*)pb;
    v8h bhi = *(const v8h*)(pb + 16);
    v16h b;
#pragma unroll
    for (int i = 0; i < 8; ++i) { b[i] = blo[i]; b[8 + i] = bhi[i]; }
    return b;
}

__global__ __launch_bounds__(256)
void k_gemm_layer(_Float16* __restrict__ hbuf, const _Float16* __restrict__ agg16,
                  const _Float16* __restrict__ wp,
                  const float* __restrict__ b0, const float* __restrict__ b1) {
    __shared__ _Float16 smem[17408];   // 128*136 halfs (weights) == 8*16*136 (C staging)

    const int tid  = threadIdx.x;
    const int lane = tid & 31;
    const int wid  = tid >> 5;
    const int g    = lane >> 4;        // lane half (0/1)
    const int ml   = lane & 15;
    const int rowBase = blockIdx.x * 128;
    const int row  = rowBase + wid * 16 + ml;

    v8f acc[8];
#pragma unroll
    for (int nt = 0; nt < 8; ++nt) {
        float b = b0[nt * 16 + ml] + b1[nt * 16 + ml];
#pragma unroll
        for (int r = 0; r < 8; ++r) acc[nt][r] = b;
    }

    for (int hop = 0; hop < 2; ++hop) {
        __syncthreads();
        const _Float16* wsrc = wp + hop * (HID * HID);
        for (int idx = tid; idx < HID * HID; idx += 256) {
            int n = idx >> 7, k = idx & 127;
            smem[n * WSTR + k] = wsrc[idx];
        }
        __syncthreads();

        const _Float16* A = (hop == 0) ? hbuf : agg16;
        const _Float16* arow = A + (size_t)row * HID;

        // prefetch A for ks = 0
        v8h alo = *(const v8h*)(arow + g * 8);
        v8h ahi = *(const v8h*)(arow + 16 + g * 8);
#pragma unroll
        for (int ks = 0; ks < HID; ks += 32) {
            v16h a;
#pragma unroll
            for (int i = 0; i < 8; ++i) { a[i] = alo[i]; a[8 + i] = ahi[i]; }
            if (ks < HID - 32) {       // prefetch next k-step
                alo = *(const v8h*)(arow + ks + 32 + g * 8);
                ahi = *(const v8h*)(arow + ks + 48 + g * 8);
            }
            v16h bcur = load_b_frag(smem, 0, ks, ml, g);
#pragma unroll
            for (int nt = 0; nt < 8; ++nt) {
                v16h bnext = bcur;
                if (nt < 7) bnext = load_b_frag(smem, nt + 1, ks, ml, g);
                acc[nt] = __builtin_amdgcn_wmma_f32_16x16x32_f16(
                    false, a, false, bcur, (short)0, acc[nt], false, false);
                bcur = bnext;
            }
        }
    }

    __syncthreads();
    // relu + stage C tiles to LDS (f16): VGPR r -> M = r + 8g, N = nt*16 + ml
#pragma unroll
    for (int nt = 0; nt < 8; ++nt)
#pragma unroll
        for (int r = 0; r < 8; ++r) {
            float v = acc[nt][r];
            v = v > 0.0f ? v : 0.0f;
            int m = r + 8 * g;
            smem[wid * (16 * WSTR) + m * WSTR + nt * 16 + ml] = (_Float16)v;
        }
    __syncthreads();

    // coalesced write-back (u32 pairs) -> in-place feature buffer
#pragma unroll
    for (int it = 0; it < 32; ++it) {
        int idx = tid + it * 256;          // u32 index over 128x128 halfs
        int r  = idx >> 6;
        int c2 = idx & 63;
        const unsigned int* ps =
            (const unsigned int*)(smem + (r >> 4) * (16 * WSTR) + (r & 15) * WSTR + c2 * 2);
        *(unsigned int*)(hbuf + (size_t)(rowBase + r) * HID + c2 * 2) = *ps;
    }
}

// ---------------------------------------------------------------------------
// Output layer (128 -> 2): one wave per node, cross-lane reduction
// ---------------------------------------------------------------------------
__global__ void k_out(const _Float16* __restrict__ h16, const _Float16* __restrict__ agg16,
                      const float* __restrict__ Wout, const float* __restrict__ bout,
                      float* __restrict__ out) {
    int node = blockIdx.x * 8 + (threadIdx.x >> 5);
    if (node >= N_NODES) return;
    int lane = threadIdx.x & 31;
    int k = lane * 4;
    v4h hv = *(const v4h*)(h16 + (size_t)node * HID + k);
    v4h av = *(const v4h*)(agg16 + (size_t)node * HID + k);
    float a0 = 0.0f, a1 = 0.0f;
#pragma unroll
    for (int j = 0; j < 4; ++j) {
        float h = (float)hv[j];
        float a = (float)av[j];
        a0 += h * Wout[(k + j) * 2 + 0] + a * Wout[HID * 2 + (k + j) * 2 + 0];
        a1 += h * Wout[(k + j) * 2 + 1] + a * Wout[HID * 2 + (k + j) * 2 + 1];
    }
#pragma unroll
    for (int off = 16; off >= 1; off >>= 1) {
        a0 += __shfl_xor(a0, off, 32);
        a1 += __shfl_xor(a1, off, 32);
    }
    if (lane == 0) {
        out[node * 2 + 0] = fmaxf(a0 + bout[0] + bout[2], 0.0f);
        out[node * 2 + 1] = fmaxf(a1 + bout[1] + bout[3], 0.0f);
    }
}

// ---------------------------------------------------------------------------
extern "C" void kernel_launch(void* const* d_in, const int* in_sizes, int n_in,
                              void* d_out, int out_size, void* d_ws, size_t ws_size,
                              hipStream_t stream) {
    const float* x_t  = (const float*)d_in[0];
    const int*   ei   = (const int*)d_in[1];
    const int*   t    = (const int*)d_in[2];
    const float* temb = (const float*)d_in[3];
    const float* Win  = (const float*)d_in[4];
    const float* bin  = (const float*)d_in[5];
    const float* Whid = (const float*)d_in[6];
    const float* bhid = (const float*)d_in[7];
    const float* Wout = (const float*)d_in[8];
    const float* bout = (const float*)d_in[9];
    float* out = (float*)d_out;

    char* ws = (char*)d_ws;                         // ~38 MB total
    float*    dinv    = (float*)(ws + 0);           // NPAD f32
    int*      cnt     = (int*)  (ws + 200192);      // NPAD i32
    int*      rowptr  = (int*)  (ws + 400384);      // NPAD+1 i32
    int*      cursor  = (int*)  (ws + 600832);      // NPAD i32
    int*      csr_src = (int*)  (ws + 801024);      // E i32
    float*    csr_w   = (float*)(ws + 7201024);     // E f32
    _Float16* h16     = (_Float16*)(ws + 13601024); // NPAD*128 f16
    _Float16* agg16   = (_Float16*)(ws + 26413312); // NPAD*128 f16
    _Float16* wp      = (_Float16*)(ws + 39225600); // 8*2*128*128 f16

    // one-time precompute: weights, features, CSR by destination
    k_cvt_weights<<<cdiv(8 * 2 * HID * HID, 256), 256, 0, stream>>>(Win, Whid, wp);
    k_embed<<<cdiv(NPAD * HID, 256), 256, 0, stream>>>(x_t, t, temb, h16);
    k_zero<<<cdiv(NPAD, 256), 256, 0, stream>>>(cnt);
    k_count<<<cdiv(N_EDGES, 256), 256, 0, stream>>>(ei, cnt);
    k_dinv<<<cdiv(NPAD, 256), 256, 0, stream>>>(cnt, dinv);
    k_scan<<<1, 1024, 0, stream>>>(cnt, rowptr);
    k_copy<<<cdiv(NPAD, 256), 256, 0, stream>>>(rowptr, cursor);
    k_fill<<<cdiv(N_EDGES, 256), 256, 0, stream>>>(ei, dinv, cursor, csr_src, csr_w);

    // 8 WMMA layers (layer 0: only 16 live input channels -> 4 lanes/node gather)
    for (int l = 0; l < 8; ++l) {
        int shift = (l == 0) ? 2 : 5;
        k_gather<<<cdiv(NPAD << shift, 256), 256, 0, stream>>>(
            rowptr, csr_src, csr_w, h16, dinv, agg16, shift);
        const float* bb0 = (l == 0) ? bin : bhid + (size_t)((l - 1) * 2) * HID;
        const float* bb1 = bb0 + HID;
        k_gemm_layer<<<NPAD / 128, 256, 0, stream>>>(
            h16, agg16, wp + (size_t)l * 2 * HID * HID, bb0, bb1);
    }

    // output layer 128 -> 2
    k_gather<<<cdiv(NPAD << 5, 256), 256, 0, stream>>>(
        rowptr, csr_src, csr_w, h16, dinv, agg16, 5);
    k_out<<<cdiv(N_NODES, 8), 256, 0, stream>>>(h16, agg16, Wout, bout, out);
}
